// CA2_82300163326041
// MI455X (gfx1250) — matrix operand order repaired and checked by reading, not compiled, not run on hardware
//
#include <hip/hip_runtime.h>

#define DD 512
#define SS 2048
#define NB 8

typedef __attribute__((ext_vector_type(16))) __bf16        v16bf;
typedef __attribute__((ext_vector_type(8)))  float         v8f;
typedef __attribute__((ext_vector_type(8)))  unsigned int  v8u;

struct U4x2  { uint4 a, b; };
struct U16x16 { unsigned short u[16]; };
struct U16x8  { unsigned short u[8]; };

static __device__ __forceinline__ v16bf bf_from2(uint4 x, uint4 y) {
    U4x2 s{x, y};
    return __builtin_bit_cast(v16bf, s);
}
static __device__ __forceinline__ v16bf bf_from8(v8u x) {
    return __builtin_bit_cast(v16bf, x);
}
// native fptrunc f32->bf16 (RNE): lets the backend use hardware cvt_pk_bf16_f32
static __device__ __forceinline__ unsigned short f2bf(float f) {
    return __builtin_bit_cast(unsigned short, (__bf16)f);
}
static __device__ __forceinline__ void put4(float* d, float4 v) {
    d[0] = v.x; d[1] = v.y; d[2] = v.z; d[3] = v.w;
}
static __device__ __forceinline__ v16bf cvt16(const float* f) {
    U16x16 s;
#pragma unroll
    for (int i = 0; i < 16; ++i) s.u[i] = f2bf(f[i]);
    return __builtin_bit_cast(v16bf, s);
}

// xor-swizzle within 16-lane row groups (wave32): and=0x1f, or=0, xor=imm
#define SWZ_F(x, imm) \
    __uint_as_float(__builtin_amdgcn_ds_swizzle(__float_as_uint(x), (((imm) << 10) | 0x1f)))

static __device__ __forceinline__ float rowmax16(float v) {
    v = fmaxf(v, SWZ_F(v, 1)); v = fmaxf(v, SWZ_F(v, 2));
    v = fmaxf(v, SWZ_F(v, 4)); v = fmaxf(v, SWZ_F(v, 8));
    return v;
}
static __device__ __forceinline__ float rowsum16(float v) {
    v += SWZ_F(v, 1); v += SWZ_F(v, 2); v += SWZ_F(v, 4); v += SWZ_F(v, 8);
    return v;
}

#define WMMA_BF16(a, b, c) \
    __builtin_amdgcn_wmma_f32_16x16x32_bf16(false, (a), false, (b), (short)0, (c), false, false)

// ---------------------------------------------------------------------------
// Kernel 1: Q1 = bf16(X @ Wxq^T + bxq), Q2 = bf16(Y @ Wyq^T + byq)
// grid = (256 row-tiles, 8 col-groups, 2 sel), block = 128 (4 waves)
// Each wave: 16 rows x 64 cols.
// ---------------------------------------------------------------------------
__global__ __launch_bounds__(128) void CA2_qproj(
    const float* __restrict__ X, const float* __restrict__ Y,
    const float* __restrict__ Wx, const float* __restrict__ bx,
    const float* __restrict__ Wy, const float* __restrict__ by,
    unsigned short* __restrict__ Q1, unsigned short* __restrict__ Q2)
{
    const int w = threadIdx.x >> 5, lane = threadIdx.x & 31;
    const int half = lane >> 4, l16 = lane & 15;
    const int sel = blockIdx.z;
    const float* A    = sel ? Y  : X;
    const float* W    = sel ? Wy : Wx;
    const float* bias = sel ? by : bx;
    unsigned short* out = sel ? Q2 : Q1;

    const int r0 = (blockIdx.x * 4 + w) * 16;   // row tile base (0..16383)
    const int n0 = blockIdx.y * 64;             // col group base

    const v8f vzero = {};
    v8f acc[4];
#pragma unroll
    for (int t = 0; t < 4; ++t) acc[t] = vzero;

    const int arow = r0 + l16;
    const float* abase = A + (size_t)arow * DD + half * 16;

    for (int kc = 0; kc < 16; ++kc) {
        float af[16];
        const float* ap = abase + kc * 32;
        put4(af + 0,  *(const float4*)(ap + 0));
        put4(af + 4,  *(const float4*)(ap + 4));
        put4(af + 8,  *(const float4*)(ap + 8));
        put4(af + 12, *(const float4*)(ap + 12));
        v16bf amat = cvt16(af);
#pragma unroll
        for (int t = 0; t < 4; ++t) {
            const float* wp = W + (size_t)(n0 + t * 16 + l16) * DD + kc * 32 + half * 16;
            float wf[16];
            put4(wf + 0,  *(const float4*)(wp + 0));
            put4(wf + 4,  *(const float4*)(wp + 4));
            put4(wf + 8,  *(const float4*)(wp + 8));
            put4(wf + 12, *(const float4*)(wp + 12));
            acc[t] = WMMA_BF16(amat, cvt16(wf), acc[t]);
        }
    }
#pragma unroll
    for (int t = 0; t < 4; ++t) {
        const float bv = bias[n0 + t * 16 + l16];
#pragma unroll
        for (int r = 0; r < 8; ++r) {
            const int row = r0 + half * 8 + r;
            out[(size_t)row * DD + n0 + t * 16 + l16] = f2bf(acc[t][r] + bv);
        }
    }
}

// ---------------------------------------------------------------------------
// Kernel 2: K_f = bf16(Qcat @ Wfk^T + bfk)  (row-major [B*S, D])
//           V_f = bf16(Qcat @ Wfv^T + bfv)  stored TRANSPOSED as VT[B, D, S]
// grid = (256, 8, 2 sel), block = 128
// ---------------------------------------------------------------------------
__global__ __launch_bounds__(128) void CA2_kvproj(
    const unsigned short* __restrict__ Q1, const unsigned short* __restrict__ Q2,
    const float* __restrict__ Wk, const float* __restrict__ bk,
    const float* __restrict__ Wv, const float* __restrict__ bv,
    unsigned short* __restrict__ Kf, unsigned short* __restrict__ VT)
{
    const int w = threadIdx.x >> 5, lane = threadIdx.x & 31;
    const int half = lane >> 4, l16 = lane & 15;
    const int sel = blockIdx.z;
    const float* W    = sel ? Wv : Wk;
    const float* bias = sel ? bv : bk;

    const int r0 = (blockIdx.x * 4 + w) * 16;
    const int n0 = blockIdx.y * 64;

    const v8f vzero = {};
    v8f acc[4];
#pragma unroll
    for (int t = 0; t < 4; ++t) acc[t] = vzero;

    const int arow = r0 + l16;

#pragma unroll 1
    for (int kc = 0; kc < 32; ++kc) {
        const unsigned short* Qs = (kc < 16) ? Q1 : Q2;
        const int kcl = kc & 15;
        const unsigned short* ap = Qs + (size_t)arow * DD + kcl * 32 + half * 16;
        v16bf amat = bf_from2(*(const uint4*)ap, *(const uint4*)(ap + 8));
#pragma unroll
        for (int t = 0; t < 4; ++t) {
            const float* wp = W + (size_t)(n0 + t * 16 + l16) * (2 * DD) + kc * 32 + half * 16;
            float wf[16];
            put4(wf + 0,  *(const float4*)(wp + 0));
            put4(wf + 4,  *(const float4*)(wp + 4));
            put4(wf + 8,  *(const float4*)(wp + 8));
            put4(wf + 12, *(const float4*)(wp + 12));
            acc[t] = WMMA_BF16(amat, cvt16(wf), acc[t]);
        }
    }

    if (sel == 0) {  // K_f row-major
#pragma unroll
        for (int t = 0; t < 4; ++t) {
            const float bvv = bias[n0 + t * 16 + l16];
#pragma unroll
            for (int r = 0; r < 8; ++r) {
                const int row = r0 + half * 8 + r;
                Kf[(size_t)row * DD + n0 + t * 16 + l16] = f2bf(acc[t][r] + bvv);
            }
        }
    } else {         // V_f transposed: VT[b][e][s], 8 consecutive s per lane
        const int batch = r0 / SS;
        const int srow  = (r0 % SS) + half * 8;
#pragma unroll
        for (int t = 0; t < 4; ++t) {
            const int e = n0 + t * 16 + l16;
            const float bvv = bias[e];
            U16x8 pk;
#pragma unroll
            for (int r = 0; r < 8; ++r) pk.u[r] = f2bf(acc[t][r] + bvv);
            *(uint4*)(VT + ((size_t)batch * DD + e) * SS + srow) =
                __builtin_bit_cast(uint4, pk);
        }
    }
}

// ---------------------------------------------------------------------------
// Kernel 3: flash attention, one 16-query tile per wave, 32 keys per step.
// out = attend(Q) + addA (+ addB if non-null).
// grid = (S/64, B), block = 128 (4 waves).
// ---------------------------------------------------------------------------
__global__ __launch_bounds__(128, 1) void CA2_attn(
    const unsigned short* __restrict__ Qb,   // [B*S, D] bf16
    const unsigned short* __restrict__ Kf,   // [B*S, D] bf16
    const unsigned short* __restrict__ VT,   // [B, D, S] bf16
    const float* __restrict__ addA,
    const float* __restrict__ addB,
    float* __restrict__ out)
{
    __shared__ __align__(32) unsigned short pst[4][16 * 32];

    const int w = threadIdx.x >> 5, lane = threadIdx.x & 31;
    const int half = lane >> 4, l16 = lane & 15;
    const int b  = blockIdx.y;
    const int q0 = blockIdx.x * 64 + w * 16;
    const size_t rowbase = (size_t)b * SS + q0;

    const v8f vzero = {};
    v8f acc[32];
#pragma unroll
    for (int t = 0; t < 32; ++t) acc[t] = vzero;
    float m[8], lsum[8];
#pragma unroll
    for (int r = 0; r < 8; ++r) { m[r] = -1e30f; lsum[r] = 0.0f; }

    const float scale = 0.044194173824159216f;  // 1/sqrt(512)
    unsigned short* pw = pst[w];
    const unsigned short* qbase  = Qb + (rowbase + l16) * DD + half * 16;
    const unsigned short* kfb    = Kf + ((size_t)b * SS + l16) * DD + half * 16;
    const unsigned short* vtbase = VT + ((size_t)b * DD + l16) * SS + half * 16;

#pragma unroll 1
    for (int kb = 0; kb < SS / 32; ++kb) {
        const int key0 = kb * 32;
        v8f c0 = vzero, c1 = vzero;
        const unsigned short* k0p = kfb + (size_t)key0 * DD;
        const unsigned short* k1p = k0p + (size_t)16 * DD;
        // prefetch next key block's K and V streams (global_prefetch_b8)
        if (kb + 1 < SS / 32) {
            __builtin_prefetch(k0p + (size_t)32 * DD, 0, 3);
            __builtin_prefetch(vtbase + key0 + 32, 0, 3);
        }
#pragma unroll
        for (int kc = 0; kc < 16; ++kc) {
            const unsigned short* qp = qbase + kc * 32;
            v16bf qa = bf_from2(*(const uint4*)qp, *(const uint4*)(qp + 8));
            v16bf b0 = bf_from8(*(const v8u*)(k0p + kc * 32));
            v16bf b1 = bf_from8(*(const v8u*)(k1p + kc * 32));
            c0 = WMMA_BF16(qa, b0, c0);
            c1 = WMMA_BF16(qa, b1, c1);
        }
        // online softmax over the 32 new keys (row = half*8 + r, cols across lanes)
#pragma unroll
        for (int r = 0; r < 8; ++r) {
            float s0 = c0[r] * scale, s1 = c1[r] * scale;
            float mx   = rowmax16(fmaxf(s0, s1));
            float mnew = fmaxf(m[r], mx);
            float p0 = __expf(s0 - mnew), p1 = __expf(s1 - mnew);
            float rs   = rowsum16(p0 + p1);
            float corr = __expf(m[r] - mnew);
            m[r] = mnew;
            lsum[r] = lsum[r] * corr + rs;
#pragma unroll
            for (int t = 0; t < 32; ++t) acc[t][r] *= corr;
            const int mrow = half * 8 + r;
            pw[mrow * 32 + l16]      = f2bf(p0);
            pw[mrow * 32 + 16 + l16] = f2bf(p1);
        }
        __builtin_amdgcn_wave_barrier();        // keep ds store->load order
        asm volatile("" ::: "memory");
        const unsigned short* pa = pw + l16 * 32 + half * 16;
        v16bf aP = bf_from2(*(const uint4*)pa, *(const uint4*)(pa + 8));
        const unsigned short* vp = vtbase + key0;
#pragma unroll
        for (int t = 0; t < 32; ++t) {
            v16bf vb = bf_from8(*(const v8u*)(vp + (size_t)t * 16 * SS));
            acc[t] = WMMA_BF16(aP, vb, acc[t]);
        }
    }

    // epilogue: normalize + residual adds
#pragma unroll
    for (int r = 0; r < 8; ++r) {
        const float inv = 1.0f / lsum[r];
        const size_t row = rowbase + half * 8 + r;
#pragma unroll
        for (int t = 0; t < 32; ++t) {
            const size_t idx = row * DD + t * 16 + l16;
            float v = acc[t][r] * inv + addA[idx];
            if (addB) v += addB[idx];
            out[idx] = v;
        }
    }
}

// ---------------------------------------------------------------------------
extern "C" void kernel_launch(void* const* d_in, const int* in_sizes, int n_in,
                              void* d_out, int out_size, void* d_ws, size_t ws_size,
                              hipStream_t stream) {
    const float* X    = (const float*)d_in[0];
    const float* Y    = (const float*)d_in[1];
    const float* W_xq = (const float*)d_in[2];
    const float* b_xq = (const float*)d_in[3];
    const float* W_yq = (const float*)d_in[4];
    const float* b_yq = (const float*)d_in[5];
    const float* W_fk = (const float*)d_in[6];
    const float* b_fk = (const float*)d_in[7];
    const float* W_fv = (const float*)d_in[8];
    const float* b_fv = (const float*)d_in[9];
    float* out = (float*)d_out;

    const size_t TOK = (size_t)NB * SS;           // 16384 tokens
    unsigned short* Q1 = (unsigned short*)d_ws;   // [TOK, D] bf16
    unsigned short* Q2 = Q1 + TOK * DD;           // [TOK, D] bf16
    unsigned short* Kf = Q2 + TOK * DD;           // [TOK, D] bf16
    unsigned short* VT = Kf + TOK * DD;           // [B, D, S] bf16  (64 MiB total)

    dim3 blk(128, 1, 1);
    CA2_qproj<<<dim3(256, 8, 2), blk, 0, stream>>>(X, Y, W_xq, b_xq, W_yq, b_yq, Q1, Q2);
    CA2_kvproj<<<dim3(256, 8, 2), blk, 0, stream>>>(Q1, Q2, W_fk, b_fk, W_fv, b_fv, Kf, VT);
    // out = attend(Q1) + X + Y
    CA2_attn<<<dim3(SS / 64, NB, 1), blk, 0, stream>>>(Q1, Kf, VT, X, Y, out);
    // out += attend(Q2)
    CA2_attn<<<dim3(SS / 64, NB, 1), blk, 0, stream>>>(Q2, Kf, VT, out, (const float*)nullptr, out);
}
